// Policy_23639499997834
// MI455X (gfx1250) — matrix operand order repaired and checked by reading, compiled, not run on hardware
//
#include <hip/hip_runtime.h>

// ---------------- problem constants ----------------
#define NN    128   // nodes
#define FIN   512   // raw feature dim
#define EE    128   // embed dim
#define H1D   256   // hidden1
#define H2D   128   // hidden2
#define MAXST 4     // steps per node

typedef __bf16        v16bf  __attribute__((ext_vector_type(16)));
typedef float         v8f    __attribute__((ext_vector_type(8)));
typedef unsigned int  u32x4  __attribute__((ext_vector_type(4)));

union FragU { v16bf v; u32x4 q[2]; };

__device__ __forceinline__ v8f wmma_bf16(v16bf a, v16bf b, v8f c) {
  return __builtin_amdgcn_wmma_f32_16x16x32_bf16(false, a, false, b, (short)0, c, false, false);
}

__device__ __forceinline__ unsigned short f2bfu(float f) {
  __bf16 h = (__bf16)f;
  return __builtin_bit_cast(unsigned short, h);
}

// ---------------- LDS layout (bytes, all 16B aligned) ----------------
#define OFF_SA   0          // s matrix   144x256 bf16  (73728)
#define OFF_HA   73728      // h1         144x256 bf16  (73728)
#define OFF_H2   147456     // h2         144x128 f32   (73728)
#define OFF_FF   221184     // feature    128x128 f32   (65536)
#define OFF_FB   286720     // feature    128x128 bf16  (32768)
#define OFF_HV   319488     // hv         512 f32       (2048)
#define OFF_LK   321536     // lk         132 f32       (528 -> pad)
#define OFF_CAND 322064     // 128 int
#define OFF_SEL  322576     // 128 int
#define OFF_UPD  323088     // 128 int
#define OFF_SCAL 323600     // 12 ints + floats
#define SMEM_BYTES 323664   // < 320KB WGP LDS

// ---------------- workspace layout (bytes) ----------------
// W1pack 65536 bf16 @0, W2pack 32768 bf16 @131072,
// WEpack 65536 bf16 @196608, FObf 65536 bf16 @327680  (total 458752 B)

// =====================================================================
// Prep: convert/pack weights into WMMA B-fragment lane order (one shot,
// fully parallel). Pack index = ((kt*NT + nt)*32 + lane)*16 + e with
// k = kt*32 + (lane>>4)*8 + (e&7) + (e>>3)*16, n = nt*16 + (lane&15).
// =====================================================================
extern "C" __global__ void policy_prep(const float* __restrict__ W1,
                                       const float* __restrict__ W2,
                                       const float* __restrict__ W_e,
                                       const float* __restrict__ FO,
                                       unsigned short* __restrict__ W1p,
                                       unsigned short* __restrict__ W2p,
                                       unsigned short* __restrict__ WEp,
                                       unsigned short* __restrict__ FOb) {
  int idx = blockIdx.x * blockDim.x + threadIdx.x;
  if (idx < 65536) {                       // W1 [256,256], NT=16, KT=8
    int e = idx & 15, l = (idx >> 4) & 31, nt = (idx >> 9) & 15, kt = idx >> 13;
    int k = kt * 32 + (l >> 4) * 8 + (e & 7) + ((e >> 3) << 4), n = nt * 16 + (l & 15);
    W1p[idx] = f2bfu(W1[k * H1D + n]);
    return;
  }
  idx -= 65536;
  if (idx < 32768) {                       // W2 [256,128], NT=8, KT=8
    int e = idx & 15, l = (idx >> 4) & 31, nt = (idx >> 9) & 7, kt = idx >> 12;
    int k = kt * 32 + (l >> 4) * 8 + (e & 7) + ((e >> 3) << 4), n = nt * 16 + (l & 15);
    W2p[idx] = f2bfu(W2[k * H2D + n]);
    return;
  }
  idx -= 32768;
  if (idx < 65536) {                       // W_e [512,128], NT=8, KT=16
    int e = idx & 15, l = (idx >> 4) & 31, nt = (idx >> 9) & 7, kt = idx >> 12;
    int k = kt * 32 + (l >> 4) * 8 + (e & 7) + ((e >> 3) << 4), n = nt * 16 + (l & 15);
    WEp[idx] = f2bfu(W_e[k * EE + n]);
    return;
  }
  idx -= 65536;
  if (idx < NN * FIN) FOb[idx] = f2bfu(FO[idx]);   // feature_origin bf16 copy
}

// =====================================================================
// Main: single persistent workgroup (8 wave32) runs the full sequential
// rollout. All WMMA in block-uniform control flow.
// =====================================================================
extern "C" __global__ void __launch_bounds__(256)
policy_main(const float* __restrict__ adj,
            const float* __restrict__ FO,
            const float* __restrict__ W_e,
            const float* __restrict__ b_e,
            const float* __restrict__ b1,
            const float* __restrict__ b2,
            const float* __restrict__ w_lk,
            const float* __restrict__ b_lk,
            const float* __restrict__ w_act,
            const float* __restrict__ b_act,
            const unsigned short* __restrict__ W1p_u,
            const unsigned short* __restrict__ W2p_u,
            const unsigned short* __restrict__ WEp_u,
            const unsigned short* __restrict__ FOb_u,
            float* __restrict__ out) {
  extern __shared__ char smem[];
  __bf16* sA  = (__bf16*)(smem + OFF_SA);
  __bf16* hA  = (__bf16*)(smem + OFF_HA);
  float*  h2  = (float*)(smem + OFF_H2);
  float*  fF  = (float*)(smem + OFF_FF);
  __bf16* fB  = (__bf16*)(smem + OFF_FB);
  float*  hv  = (float*)(smem + OFF_HV);
  float*  lk  = (float*)(smem + OFF_LK);
  int*    cand = (int*)(smem + OFF_CAND);
  int*    sel  = (int*)(smem + OFF_SEL);
  int*    upd  = (int*)(smem + OFF_UPD);
  int*    isc  = (int*)(smem + OFF_SCAL);          // 0:ut 1:at 2:skip 3:done 4:cnt 5..8:selList 9:utc
  float*  fsc  = (float*)(smem + OFF_SCAL + 48);   // 0: 1/denominator

  const __bf16* W1p = (const __bf16*)W1p_u;
  const __bf16* W2p = (const __bf16*)W2p_u;
  const __bf16* WEp = (const __bf16*)WEp_u;
  const __bf16* FOb = (const __bf16*)FOb_u;

  const int tid  = threadIdx.x;
  const int lane = tid & 31;
  const int wave = tid >> 5;
  const int lr   = lane & 15;   // row (A) / col (B,D) within tile
  const int lh   = lane >> 4;   // half selector

  // ---- Phase 0: feature0 = relu(FO @ W_e + b_e), M=128 N=128 K=512 ----
  for (int t = wave; t < 64; t += 8) {
    const int mt = t >> 3, nt = t & 7;
    v8f acc = {};
    for (int kt = 0; kt < 16; ++kt) {
      FragU a, b;
      const __bf16* ap = FOb + (mt * 16 + lr) * FIN + kt * 32 + lh * 8;
      a.q[0] = *(const u32x4*)ap;
      a.q[1] = *(const u32x4*)(ap + 16);
      const __bf16* bp = WEp + (((kt * 8 + nt) * 32 + lane) << 4);
      b.q[0] = *(const u32x4*)bp;
      b.q[1] = *(const u32x4*)(bp + 8);
      acc = wmma_bf16(a.v, b.v, acc);
    }
    const int   col  = nt * 16 + lr;
    const float bias = b_e[col];
    const int   rb   = mt * 16 + lh * 8;
    for (int r = 0; r < 8; ++r) {
      float v = acc[r] + bias; v = v > 0.f ? v : 0.f;
      fF[(rb + r) * EE + col] = v;
      fB[(rb + r) * EE + col] = (__bf16)v;
    }
  }
  if (tid < NN) upd[tid] = 0;
  if (tid == 0) isc[3] = 0;
  __syncthreads();

  // ---- sequential node loop ----
  for (int i = 0; i < NN; ++i) {
    if (tid < NN) {
      cand[tid] = (adj[i * NN + tid] != 0.f) ? 1 : 0;
      sel[tid]  = 0;
    }
    if (tid == 0) isc[3] = 0;   // done = false
    __syncthreads();

    for (int st = 0; st < MAXST; ++st) {
      const int done = isc[3];  // block-uniform (read after barrier)
      if (!done) {
        // (1) build s = [x_i | nbr] rows 0..143 (pad), quirk: nbr row 0 zeroed
        for (int row = tid; row < 144; row += 256) {
          u32x4*       dst = (u32x4*)(sA + row * (2 * EE));
          const u32x4  z   = {0, 0, 0, 0};
          const u32x4* xi  = (const u32x4*)(fB + i * EE);
          if (row < NN) {
            for (int c = 0; c < 16; ++c) dst[c] = xi[c];
            if (row > 0) {
              const u32x4* nb = (const u32x4*)(fB + row * EE);
              for (int c = 0; c < 16; ++c) dst[16 + c] = nb[c];
            } else {
              for (int c = 0; c < 16; ++c) dst[16 + c] = z;
            }
          } else if (row == NN) {          // sentinel [x_i | 0]
            for (int c = 0; c < 16; ++c) dst[c] = xi[c];
            for (int c = 0; c < 16; ++c) dst[16 + c] = z;
          } else {                         // M padding
            for (int c = 0; c < 32; ++c) dst[c] = z;
          }
        }
        __syncthreads();

        // (2) h1 = relu(s @ W1 + b1)   M=144 N=256 K=256   (144 tiles / 8 waves)
        for (int t = wave; t < 144; t += 8) {
          const int mt = t >> 4, nt = t & 15;
          v8f acc = {};
          for (int kt = 0; kt < 8; ++kt) {
            FragU a, b;
            const __bf16* ap = sA + (mt * 16 + lr) * (2 * EE) + kt * 32 + lh * 8;
            a.q[0] = *(const u32x4*)ap;
            a.q[1] = *(const u32x4*)(ap + 16);
            const __bf16* bp = W1p + (((kt * 16 + nt) * 32 + lane) << 4);
            b.q[0] = *(const u32x4*)bp;
            b.q[1] = *(const u32x4*)(bp + 8);
            acc = wmma_bf16(a.v, b.v, acc);
          }
          const int   col  = nt * 16 + lr;
          const float bias = b1[col];
          const int   rb   = mt * 16 + lh * 8;
          for (int r = 0; r < 8; ++r) {
            float v = acc[r] + bias; v = v > 0.f ? v : 0.f;
            hA[(rb + r) * H1D + col] = (__bf16)v;
          }
        }
        __syncthreads();

        // (3) h2 = relu(h1 @ W2 + b2)  M=144 N=128 K=256   (72 tiles / 8 waves)
        for (int t = wave; t < 72; t += 8) {
          const int mt = t >> 3, nt = t & 7;
          v8f acc = {};
          for (int kt = 0; kt < 8; ++kt) {
            FragU a, b;
            const __bf16* ap = hA + (mt * 16 + lr) * H1D + kt * 32 + lh * 8;
            a.q[0] = *(const u32x4*)ap;
            a.q[1] = *(const u32x4*)(ap + 16);
            const __bf16* bp = W2p + (((kt * 8 + nt) * 32 + lane) << 4);
            b.q[0] = *(const u32x4*)bp;
            b.q[1] = *(const u32x4*)(bp + 8);
            acc = wmma_bf16(a.v, b.v, acc);
          }
          const int   col  = nt * 16 + lr;
          const float bias = b2[col];
          const int   rb   = mt * 16 + lh * 8;
          for (int r = 0; r < 8; ++r) {
            float v = acc[r] + bias; v = v > 0.f ? v : 0.f;
            h2[(rb + r) * H2D + col] = v;
          }
        }
        __syncthreads();

        // (4) lk head, masked by candidates (+ sentinel always valid), f32
        if (tid < NN + 1) {
          float s = b_lk[0];
          const float* hr = h2 + tid * H2D;
          for (int k = 0; k < H2D; ++k) s += hr[k] * w_lk[k];
          const bool valid = (tid == NN) ? true : (cand[tid] != 0);
          lk[tid] = valid ? s : -__builtin_huge_valf();
        }
        __syncthreads();

        // (5) argmax (first-max, matching jnp.argmax), act head, bookkeeping
        if (tid == 0) {
          int ut = 0; float best = lk[0];
          for (int v = 1; v <= NN; ++v) { if (lk[v] > best) { best = lk[v]; ut = v; } }
          const int is_end = (ut == NN) ? 1 : 0;
          const int utc    = ut < (NN - 1) ? ut : (NN - 1);
          const float* hr = h2 + utc * H2D;
          float a0 = b_act[0], a1 = b_act[1];
          for (int k = 0; k < H2D; ++k) { a0 += hr[k] * w_act[2 * k]; a1 += hr[k] * w_act[2 * k + 1]; }
          const int at = (a1 > a0) ? 1 : 0;  // argmax==1; tie -> index 0
          int cnt = 0;
          for (int v = 0; v < NN; ++v) { if (sel[v]) { isc[5 + cnt] = v; ++cnt; } }
          isc[0] = ut; isc[1] = at; isc[2] = is_end; isc[4] = cnt; isc[9] = utc;
          fsc[0] = 1.0f / ((float)cnt + (float)at + 1.0f);
          if (is_end) isc[3] = 1;            // done for remaining steps
        }
        __syncthreads();

        const int skip = isc[2];             // block-uniform
        if (!skip) {
          const int   utc = isc[9], at = isc[1], cnt = isc[4];
          const float inv = fsc[0];
          // hv = (sel_sum + at*x_ut + x_self) / (cnt + at + 1)   (f32)
          for (int f = tid; f < FIN; f += 256) {
            float s = FO[i * FIN + f] + (at ? FO[utc * FIN + f] : 0.f);
            for (int j = 0; j < cnt; ++j) s += FO[isc[5 + j] * FIN + f];
            hv[f] = s * inv;
          }
          __syncthreads();
          // embed(hv) on threads 0..127, embed(x_utc) on threads 128..255 (f32)
          const int  o  = tid & (EE - 1);
          const bool g0 = tid < EE;
          float s = b_e[o];
          if (g0) {
            for (int f = 0; f < FIN; ++f) s += hv[f] * W_e[f * EE + o];
          } else {
            const float* xu = FO + utc * FIN;
            for (int f = 0; f < FIN; ++f) s += xu[f] * W_e[f * EE + o];
          }
          s = s > 0.f ? s : 0.f;
          if (g0) { fF[i * EE + o] = s; fB[i * EE + o] = (__bf16)s; }
          __syncthreads();   // row utc write must win when utc == i
          if (!g0) { fF[utc * EE + o] = s; fB[utc * EE + o] = (__bf16)s; }
          if (tid == 0) {
            upd[i] = 1; upd[utc] = 1;
            if (at) sel[utc] = 1;
            cand[utc] = 0;
          }
        }
        __syncthreads();
      } // !done
    }   // steps
    __syncthreads();
  }     // nodes

  // ---- output: feature masked by updated ----
  for (int idx = tid; idx < NN * EE; idx += 256) {
    out[idx] = upd[idx >> 7] ? fF[idx] : 0.f;
  }
}

// =====================================================================
extern "C" void kernel_launch(void* const* d_in, const int* in_sizes, int n_in,
                              void* d_out, int out_size, void* d_ws, size_t ws_size,
                              hipStream_t stream) {
  (void)in_sizes; (void)n_in; (void)out_size; (void)ws_size;
  const float* adj   = (const float*)d_in[0];
  const float* FO    = (const float*)d_in[1];
  /* d_in[2] = labels (unused in forward) */
  const float* W_e   = (const float*)d_in[3];
  const float* b_e   = (const float*)d_in[4];
  const float* W1    = (const float*)d_in[5];
  const float* b1    = (const float*)d_in[6];
  const float* W2    = (const float*)d_in[7];
  const float* b2    = (const float*)d_in[8];
  const float* w_lk  = (const float*)d_in[9];
  const float* b_lk  = (const float*)d_in[10];
  const float* w_act = (const float*)d_in[11];
  const float* b_act = (const float*)d_in[12];

  char* ws = (char*)d_ws;
  unsigned short* W1p = (unsigned short*)(ws + 0);        // 65536 bf16
  unsigned short* W2p = (unsigned short*)(ws + 131072);   // 32768 bf16
  unsigned short* WEp = (unsigned short*)(ws + 196608);   // 65536 bf16
  unsigned short* FOb = (unsigned short*)(ws + 327680);   // 65536 bf16

  const int prepTotal = 65536 + 32768 + 65536 + 65536;
  policy_prep<<<(prepTotal + 255) / 256, 256, 0, stream>>>(W1, W2, W_e, FO,
                                                           W1p, W2p, WEp, FOb);
  policy_main<<<1, 256, SMEM_BYTES, stream>>>(adj, FO, W_e, b_e, b1, b2,
                                              w_lk, b_lk, w_act, b_act,
                                              W1p, W2p, WEp, FOb, (float*)d_out);
}
